// NEP3Model_4432406249565
// MI455X (gfx1250) — compile-verified, matrix operand order checked by reading, not run on hardware
//
#include <hip/hip_runtime.h>
#include <math.h>

typedef _Float16 h16 __attribute__((ext_vector_type(16)));
typedef _Float16 h8v __attribute__((ext_vector_type(8)));
typedef float    f8v __attribute__((ext_vector_type(8)));

#define NTYPES   2
#define KR       8      // radial Chebyshev order
#define KA       8      // angular Chebyshev order
#define NR       5      // NMAX_R + 1
#define NA       5      // NMAX_A + 1
#define RC_R     8.0f
#define RC_A     6.0f
#define NSPH     24     // spherical harmonic components l=1..4
#define DESC     25     // 5 radial + 20 angular
#define KPAD     32     // K padded for WMMA 16x16x32
#define NEUR     100
#define NEUR_PAD 112    // 7 tiles of 16
#define PI_F     3.14159265358979f

// Branch-free tanh: native V_TANH_F32 if the toolchain exposes it, else
// exp2/rcp-based (v_exp_f32 + v_rcp_f32), no divergence, no spills.
__device__ __forceinline__ float fast_tanh(float x) {
#if __has_builtin(__builtin_amdgcn_tanhf)
    return __builtin_amdgcn_tanhf(x);
#else
    float ax = fabsf(x);
    // 2*log2(e) = 2.885390081777927
    float e2 = __builtin_amdgcn_exp2f(ax * 2.8853900817779268f);   // exp(2|x|)
    float t  = 1.0f - 2.0f * __builtin_amdgcn_rcpf(e2 + 1.0f);     // |x| big -> 1
    return copysignf(t, x);
#endif
}

// ---------------------------------------------------------------- utilities

__global__ void k_zero(float* __restrict__ p, int n) {
    int i = blockIdx.x * blockDim.x + threadIdx.x;
    if (i < n) p[i] = 0.0f;
}

// Pack W1 (100x25 f32, row-major) into padded f16 (112x32), zeros elsewhere.
__global__ void k_packW1(const float* __restrict__ W1, _Float16* __restrict__ W1h) {
    int t = blockIdx.x * blockDim.x + threadIdx.x;
    if (t < NEUR_PAD * KPAD) {
        int nrn = t / KPAD, k = t % KPAD;
        float v = (nrn < NEUR && k < DESC) ? W1[nrn * DESC + k] : 0.0f;
        W1h[t] = (_Float16)v;
    }
}

// Chebyshev radial basis: basis[k] for k = 0..8. Argument of cos is in
// [0, pi] (edges guarantee r < rc), so the fast native path is safe.
__device__ __forceinline__ void cheb9(float r, float rc, float* basis) {
    float rcinv = 1.0f / rc;
    float fc = (r <= rc) ? 0.5f * (__cosf(PI_F * r * rcinv) + 1.0f) : 0.0f;
    float t  = r * rcinv - 1.0f;
    float x  = 2.0f * t * t - 1.0f;
    float hfc = 0.5f * fc;
    basis[0] = fc;
    basis[1] = (x + 1.0f) * hfc;
    float Tm2 = 1.0f, Tm1 = x;
#pragma unroll
    for (int k = 2; k <= 8; ++k) {
        float Tm = 2.0f * x * Tm1 - Tm2;
        basis[k] = (Tm + 1.0f) * hfc;
        Tm2 = Tm1; Tm1 = Tm;
    }
}

// ---------------------------------------------------------------- radial edges

__global__ void k_radial(const float* __restrict__ pos, const int* __restrict__ types,
                         const int* __restrict__ ei, const int* __restrict__ ej,
                         const float* __restrict__ c_rad, float* __restrict__ q_rad,
                         int nedge) {
    int e = blockIdx.x * blockDim.x + threadIdx.x;
    if (e >= nedge) return;
    int i = ei[e], j = ej[e];
    float dx = pos[3 * j + 0] - pos[3 * i + 0];
    float dy = pos[3 * j + 1] - pos[3 * i + 1];
    float dz = pos[3 * j + 2] - pos[3 * i + 2];
    float d = sqrtf(dx * dx + dy * dy + dz * dz);
    float b[KR + 1];
    cheb9(d, RC_R, b);
    const float* c = c_rad + (types[i] * NTYPES + types[j]) * NR * (KR + 1);
#pragma unroll
    for (int n = 0; n < NR; ++n) {
        float g = 0.0f;
#pragma unroll
        for (int k = 0; k <= KR; ++k) g += b[k] * c[n * (KR + 1) + k];
        atomicAdd(&q_rad[i * NR + n], g);
    }
}

// ---------------------------------------------------------------- angular edges

__global__ void k_angular(const float* __restrict__ pos, const int* __restrict__ types,
                          const int* __restrict__ ei, const int* __restrict__ ej,
                          const float* __restrict__ c_ang, float* __restrict__ S,
                          int nedge) {
    int e = blockIdx.x * blockDim.x + threadIdx.x;
    if (e >= nedge) return;
    int i = ei[e], j = ej[e];
    float dx = pos[3 * j + 0] - pos[3 * i + 0];
    float dy = pos[3 * j + 1] - pos[3 * i + 1];
    float dz = pos[3 * j + 2] - pos[3 * i + 2];
    float d = sqrtf(dx * dx + dy * dy + dz * dz);
    float b[KA + 1];
    cheb9(d, RC_A, b);
    const float* c = c_ang + (types[i] * NTYPES + types[j]) * NA * (KA + 1);
    float g[NA];
#pragma unroll
    for (int n = 0; n < NA; ++n) {
        float acc = 0.0f;
#pragma unroll
        for (int k = 0; k <= KA; ++k) acc += b[k] * c[n * (KA + 1) + k];
        g[n] = acc;
    }
    float dinv = 1.0f / d;
    float x = dx * dinv, y = dy * dinv, z = dz * dinv;
    float x2 = x * x, y2 = y * y, z2 = z * z;
    float Y[NSPH];
    // l = 1
    Y[0]  = 0.48860251f * x;
    Y[1]  = 0.48860251f * y;
    Y[2]  = 0.48860251f * z;
    // l = 2
    Y[3]  = 1.09254843f * x * y;
    Y[4]  = 1.09254843f * y * z;
    Y[5]  = 0.31539157f * (3.0f * z2 - 1.0f);
    Y[6]  = 1.09254843f * x * z;
    Y[7]  = 0.54627422f * (x2 - y2);
    // l = 3
    Y[8]  = 0.59004359f * y * (3.0f * x2 - y2);
    Y[9]  = 2.89061144f * x * y * z;
    Y[10] = 0.45704580f * y * (5.0f * z2 - 1.0f);
    Y[11] = 0.37317633f * (5.0f * z2 * z - 3.0f * z);
    Y[12] = 0.45704580f * x * (5.0f * z2 - 1.0f);
    Y[13] = 1.44530572f * z * (x2 - y2);
    Y[14] = 0.59004359f * x * (x2 - 3.0f * y2);
    // l = 4
    Y[15] = 2.50334294f * x * y * (x2 - y2);
    Y[16] = 1.77013077f * y * z * (3.0f * x2 - y2);
    Y[17] = 0.94617470f * x * y * (7.0f * z2 - 1.0f);
    Y[18] = 0.66904654f * y * z * (7.0f * z2 - 3.0f);
    Y[19] = 0.10578555f * (35.0f * z2 * z2 - 30.0f * z2 + 3.0f);
    Y[20] = 0.66904654f * x * z * (7.0f * z2 - 3.0f);
    Y[21] = 0.47308735f * (x2 - y2) * (7.0f * z2 - 1.0f);
    Y[22] = 1.77013077f * x * z * (x2 - 3.0f * y2);
    Y[23] = 0.62583574f * (x2 * (x2 - 3.0f * y2) - y2 * (3.0f * x2 - y2));

    float* Srow = S + (size_t)i * (NA * NSPH);
#pragma unroll
    for (int n = 0; n < NA; ++n) {
        float gn = g[n];
#pragma unroll
        for (int m = 0; m < NSPH; ++m)
            atomicAdd(&Srow[n * NSPH + m], gn * Y[m]);
    }
}

// ---------------------------------------------------------------- descriptors

__global__ void k_desc(const float* __restrict__ q_rad, const float* __restrict__ S,
                       _Float16* __restrict__ desc_h, int natoms) {
    int a = blockIdx.x * blockDim.x + threadIdx.x;
    if (a >= natoms) return;
    _Float16* dr = desc_h + (size_t)a * KPAD;
#pragma unroll
    for (int n = 0; n < NR; ++n) dr[n] = (_Float16)q_rad[a * NR + n];
    const int offs[5] = {0, 3, 8, 15, 24};
    const float fourpi = 12.566370614f;
    const float* Srow = S + (size_t)a * (NA * NSPH);
#pragma unroll
    for (int li = 0; li < 4; ++li) {
        float fac = fourpi / (float)(2 * (li + 1) + 1);
#pragma unroll
        for (int n = 0; n < NA; ++n) {
            float s = 0.0f;
            for (int m = offs[li]; m < offs[li + 1]; ++m) {
                float v = Srow[n * NSPH + m];
                s += v * v;
            }
            dr[NR + li * NA + n] = (_Float16)(fac * s);
        }
    }
#pragma unroll
    for (int k = DESC; k < KPAD; ++k) dr[k] = (_Float16)0.0f;
}

// ---------------------------------------------------------------- WMMA MLP
// One wave handles a 16-atom M-tile. A = desc (16x32 f16, K padded),
// B = W1 tile (32x16 f16: 16 neurons), single K-step per tile (K=32 covers 25).
// 7 N-tiles cover 112 >= 100 neurons; padding masked through W2.

__global__ void k_mlp(const _Float16* __restrict__ desc_h, const _Float16* __restrict__ W1h,
                      const float* __restrict__ b1, const float* __restrict__ W2,
                      const float* __restrict__ b2, const float* __restrict__ aE,
                      const int* __restrict__ types, float* __restrict__ e_out) {
    int lane  = threadIdx.x & 31;
    int wave  = threadIdx.x >> 5;
    int mtile = blockIdx.x * (blockDim.x >> 5) + wave;

    // A-matrix layout (16-bit A, 16x32): lanes 0-15 -> M=lane, K in {0..7,16..23};
    // lanes 16-31 -> M=lane-16, K in {8..15,24..31}.
    int row   = mtile * 16 + (lane & 15);
    int kbase = (lane < 16) ? 0 : 8;
    const h8v* pa0 = (const h8v*)(desc_h + (size_t)row * KPAD + kbase);
    const h8v* pa1 = (const h8v*)(desc_h + (size_t)row * KPAD + kbase + 16);
    h8v alo = *pa0, ahi = *pa1;
    h16 A;
#pragma unroll
    for (int q = 0; q < 8; ++q) { A[q] = alo[q]; A[8 + q] = ahi[q]; }

    float esum[8];
#pragma unroll
    for (int r = 0; r < 8; ++r) esum[r] = 0.0f;

#pragma unroll
    for (int nt = 0; nt < 7; ++nt) {
        // B-matrix 32x16: lane n in 0-15 -> column n, K in {0..7,16..23};
        // lane n+16 -> column n, K in {8..15,24..31}. B[k][n] = W1h[n][k].
        int nrn = nt * 16 + (lane & 15);
        const h8v* pb0 = (const h8v*)(W1h + (size_t)nrn * KPAD + kbase);
        const h8v* pb1 = (const h8v*)(W1h + (size_t)nrn * KPAD + kbase + 16);
        h8v blo = *pb0, bhi = *pb1;
        h16 B;
#pragma unroll
        for (int q = 0; q < 8; ++q) { B[q] = blo[q]; B[8 + q] = bhi[q]; }

        f8v C = {};
        C = __builtin_amdgcn_wmma_f32_16x16x32_f16(false, A, false, B,
                                                   (short)0, C, false, false);

        float b1v = (nrn < NEUR) ? b1[nrn] : 0.0f;
        float w2v = (nrn < NEUR) ? W2[nrn] : 0.0f;
        // C/D layout: VGPR r, lanes 0-15 -> (M=r, N=lane); lanes 16-31 -> (M=r+8).
#pragma unroll
        for (int r = 0; r < 8; ++r) {
            float h = fast_tanh(C[r] - b1v);
            esum[r] += h * w2v;
        }
    }

    // Reduce over the 16 lanes (N dimension) within each half-wave.
#pragma unroll
    for (int r = 0; r < 8; ++r) {
#pragma unroll
        for (int off = 1; off < 16; off <<= 1)
            esum[r] += __shfl_xor(esum[r], off, 32);
    }

    if ((lane & 15) == 0) {
        int rbase = mtile * 16 + ((lane == 16) ? 8 : 0);
        float b2v = b2[0];
#pragma unroll
        for (int r = 0; r < 8; ++r) {
            int atom = rbase + r;
            e_out[atom] = esum[r] - b2v + aE[types[atom]];
        }
    }
}

// ---------------------------------------------------------------- launch

extern "C" void kernel_launch(void* const* d_in, const int* in_sizes, int n_in,
                              void* d_out, int out_size, void* d_ws, size_t ws_size,
                              hipStream_t stream) {
    const float* pos   = (const float*)d_in[0];
    const float* c_rad = (const float*)d_in[1];
    const float* c_ang = (const float*)d_in[2];
    const float* W1    = (const float*)d_in[3];
    const float* b1    = (const float*)d_in[4];
    const float* W2    = (const float*)d_in[5];
    const float* b2    = (const float*)d_in[6];
    const float* aE    = (const float*)d_in[7];
    const int*   types = (const int*)d_in[8];
    const int*   ei_r  = (const int*)d_in[9];
    const int*   ej_r  = (const int*)d_in[10];
    const int*   ei_a  = (const int*)d_in[11];
    const int*   ej_a  = (const int*)d_in[12];

    int natoms = in_sizes[8];
    int er = in_sizes[9];
    int ea = in_sizes[11];

    // Workspace layout (all naturally aligned):
    float*    q_rad  = (float*)d_ws;                           // natoms * 5 f32
    float*    S      = q_rad + (size_t)natoms * NR;            // natoms * 120 f32
    _Float16* desc_h = (_Float16*)(S + (size_t)natoms * (NA * NSPH)); // natoms*32 f16
    _Float16* W1h    = desc_h + (size_t)natoms * KPAD;         // 112*32 f16

    int zn = natoms * (NR + NA * NSPH);   // zero q_rad + S
    k_zero  <<<(zn + 255) / 256, 256, 0, stream>>>(q_rad, zn);
    k_packW1<<<(NEUR_PAD * KPAD + 255) / 256, 256, 0, stream>>>(W1, W1h);
    k_radial <<<(er + 255) / 256, 256, 0, stream>>>(pos, types, ei_r, ej_r, c_rad, q_rad, er);
    k_angular<<<(ea + 255) / 256, 256, 0, stream>>>(pos, types, ei_a, ej_a, c_ang, S, ea);
    k_desc   <<<(natoms + 255) / 256, 256, 0, stream>>>(q_rad, S, desc_h, natoms);

    int mtiles = natoms / 16;             // 1024 for 16384 atoms
    k_mlp<<<mtiles / 4, 128, 0, stream>>>(desc_h, W1h, b1, W2, b2, aE, types, (float*)d_out);
}
